// VisionModel_29738353557831
// MI455X (gfx1250) — compile-verified, matrix operand order checked by reading, not run on hardware
//
#include <hip/hip_runtime.h>
#include <hip/hip_bf16.h>
#include <stdint.h>

// ---------------- types ----------------
typedef __hip_bfloat16 bfst;                       // 16-bit storage type
typedef __bf16 bf16t;                              // clang bf16 for WMMA
typedef bf16t v16bf __attribute__((ext_vector_type(16)));
typedef bf16t v8bf  __attribute__((ext_vector_type(8)));
typedef float v8f   __attribute__((ext_vector_type(8)));

// ---------------- model constants (GRID is static in the reference) ----------
#define DMODEL 1152
#define NHEAD  16
#define KHEAD  72
#define KPADH  96      // head dim padded to mult of 32 for WMMA K
#define NTOK   6272
#define IDIM   4304
#define IPAD   4352    // mult of 64
#define MDIM   4608
#define OUTD   2048
#define CIN    1536
#define QKVD   3456
#define PPAD   16      // LDS row padding (halfs) to de-conflict banks

// ---------------- helpers ----------------
__device__ __forceinline__ bf16t zbf() {
  unsigned short z = 0; bf16t r; __builtin_memcpy(&r, &z, 2); return r;
}
__device__ __forceinline__ float b2f(bfst x) { return __bfloat162float(x); }

__device__ __forceinline__ v8bf ld8(const bfst* p) {
  return *reinterpret_cast<const v8bf*>(p);
}
__device__ __forceinline__ v16bf mk16(v8bf lo, v8bf hi) {
  v16bf r;
#pragma unroll
  for (int i = 0; i < 8; ++i) { r[i] = lo[i]; r[8 + i] = hi[i]; }
  return r;
}
__device__ __forceinline__ v8f wmma_bf(v16bf a, v16bf b, v8f c) {
  return __builtin_amdgcn_wmma_f32_16x16x32_bf16(false, a, false, b, (short)0, c,
                                                 false, false);
}
__device__ __forceinline__ float gelu_f(float x) {
  return 0.5f * x * (1.f + tanhf(0.7978845608028654f * (x + 0.044715f * x * x * x)));
}
__device__ __forceinline__ float wred32(float v) {
#pragma unroll
  for (int m = 16; m > 0; m >>= 1) v += __shfl_xor(v, m, 32);
  return v;
}
__device__ __forceinline__ float wmax32(float v) {
#pragma unroll
  for (int m = 16; m > 0; m >>= 1) v = fmaxf(v, __shfl_xor(v, m, 32));
  return v;
}
// async copy 16B global -> LDS (gfx1250 ASYNCcnt path), plus +16B variant
__device__ __forceinline__ void async_g2l_32B(const bfst* gsrc, bfst* ldst) {
  unsigned lo = (unsigned)(uintptr_t)(void*)ldst;          // low 32b = LDS offset
  unsigned long long ga = (unsigned long long)(uintptr_t)(const void*)gsrc;
  asm volatile("global_load_async_to_lds_b128 %0, %1, off" ::"v"(lo), "v"(ga)
               : "memory");
  asm volatile("global_load_async_to_lds_b128 %0, %1, off offset:16" ::"v"(lo),
               "v"(ga)
               : "memory");
}
__device__ __forceinline__ void wait_async0() {
  asm volatile("s_wait_asynccnt 0x0" ::: "memory");
}
// token -> (row,col,grid h,grid w); frame-expanded static segments
__device__ __forceinline__ void tok2rc(int t, int& r, int& c, int& hh, int& ww) {
  int idx;
  if (t < 2304)      { idx = t;        hh = 48; ww = 48; }
  else if (t < 4352) { idx = t - 2304; hh = 32; ww = 64; }
  else if (t < 5312) { idx = t - 4352; hh = 24; ww = 40; }
  else               { idx = t - 5312; hh = 24; ww = 40; }
  int mw = ww >> 1;
  int j = idx & 1, i = (idx >> 1) & 1, blk = idx >> 2;
  int bw = blk % mw, bh = blk / mw;
  r = (bh << 1) | i;  c = (bw << 1) | j;
}

// ---------------- elementwise / prep kernels ----------------
__global__ void f2bf_kernel(const float* __restrict__ src, bfst* __restrict__ dst,
                            size_t n) {
  size_t i = (size_t)blockIdx.x * blockDim.x + threadIdx.x;
  if (i < n) dst[i] = __float2bfloat16(src[i]);
}

// W[Kd][N] f32 -> Wt[N][Kpad] bf16 (zero-fill K padding)
__global__ void wt_kernel(const float* __restrict__ W, bfst* __restrict__ Wt,
                          int Kd, int N, int Kpad) {
  size_t i = (size_t)blockIdx.x * blockDim.x + threadIdx.x;
  size_t tot = (size_t)N * Kpad;
  if (i >= tot) return;
  int n = (int)(i / Kpad), k = (int)(i % Kpad);
  Wt[i] = (k < Kd) ? __float2bfloat16(W[(size_t)k * N + n]) : __float2bfloat16(0.f);
}

// bilinear pos-embed add (static grid)
__global__ void posadd_kernel(const float* __restrict__ pos_table,
                              float* __restrict__ X) {
  int t = blockIdx.x;
  int r, c, hh, ww; tok2rc(t, r, c, hh, ww);
  float hi = (float)r * 47.f / (float)(hh - 1);
  float wi = (float)c * 47.f / (float)(ww - 1);
  int hf = (int)floorf(hi), wf = (int)floorf(wi);
  int hc = min(hf + 1, 47), wc = min(wf + 1, 47);
  float dh = hi - (float)hf, dw = wi - (float)wf;
  float w00 = (1.f - dh) * (1.f - dw), w01 = (1.f - dh) * dw;
  float w10 = dh * (1.f - dw), w11 = dh * dw;
  const float* p00 = pos_table + (size_t)(hf * 48 + wf) * DMODEL;
  const float* p01 = pos_table + (size_t)(hf * 48 + wc) * DMODEL;
  const float* p10 = pos_table + (size_t)(hc * 48 + wf) * DMODEL;
  const float* p11 = pos_table + (size_t)(hc * 48 + wc) * DMODEL;
  float* xr = X + (size_t)t * DMODEL;
  for (int j = threadIdx.x; j < DMODEL; j += blockDim.x)
    xr[j] += p00[j] * w00 + p01[j] * w01 + p10[j] * w10 + p11[j] * w11;
}

__global__ void rope_tab_kernel(float* __restrict__ cosT, float* __restrict__ sinT) {
  int t = blockIdx.x, u = threadIdx.x;
  if (u >= 36) return;
  int r, c, hh, ww; tok2rc(t, r, c, hh, ww);
  int uu = (u < 18) ? u : u - 18;
  float pos = (u < 18) ? (float)r : (float)c;
  float inv = powf(10000.f, -(2.f * (float)uu) / 36.f);
  float ang = pos * inv;
  float cv = cosf(ang), sv = sinf(ang);
  cosT[t * KHEAD + u] = cv; cosT[t * KHEAD + 36 + u] = cv;
  sinT[t * KHEAD + u] = sv; sinT[t * KHEAD + 36 + u] = sv;
}

// layernorm (fp32 in) -> bf16 out; one wave per token (wave32)
__global__ void ln_kernel(const float* __restrict__ X, const float* __restrict__ s,
                          const float* __restrict__ b, bfst* __restrict__ out,
                          int ldo) {
  int token = blockIdx.x * 4 + (threadIdx.x >> 5);
  int lane = threadIdx.x & 31;
  if (token >= NTOK) return;
  const float* row = X + (size_t)token * DMODEL;
  float sum = 0.f, sq = 0.f;
  for (int j = lane; j < DMODEL; j += 32) { float v = row[j]; sum += v; sq += v * v; }
  sum = wred32(sum); sq = wred32(sq);
  float mu = sum * (1.f / DMODEL);
  float var = sq * (1.f / DMODEL) - mu * mu;
  float inv = rsqrtf(var + 1e-6f);
  bfst* orow = out + (size_t)token * ldo;
  for (int j = lane; j < DMODEL; j += 32)
    orow[j] = __float2bfloat16((row[j] - mu) * inv * s[j] + b[j]);
}

// qkv(bf16, token-major 3456) -> rope'd q,k (head-padded to 96) + transposed v
__global__ void rope_apply_kernel(const bfst* __restrict__ qkv,
                                  const float* __restrict__ cosT,
                                  const float* __restrict__ sinT,
                                  bfst* __restrict__ qp, bfst* __restrict__ kp,
                                  bfst* __restrict__ vT) {
  int t = blockIdx.x;
  const bfst* row = qkv + (size_t)t * QKVD;
  const float* ct = cosT + t * KHEAD;
  const float* st = sinT + t * KHEAD;
  const float scale = rsqrtf((float)KHEAD);
  for (int x = threadIdx.x; x < NHEAD * KPADH; x += blockDim.x) {
    int h = x / KPADH, f = x % KPADH;
    float qv = 0.f, kv = 0.f;
    if (f < KHEAD) {
      float cosv = ct[f], sinv = st[f];
      int fr = (f < 36) ? f + 36 : f - 36;
      float q = b2f(row[h * KHEAD + f]);
      float qr = b2f(row[h * KHEAD + fr]); if (f < 36) qr = -qr;
      float k = b2f(row[DMODEL + h * KHEAD + f]);
      float kr = b2f(row[DMODEL + h * KHEAD + fr]); if (f < 36) kr = -kr;
      qv = (q * cosv + qr * sinv) * scale;
      kv = k * cosv + kr * sinv;
    }
    qp[(size_t)t * (NHEAD * KPADH) + x] = __float2bfloat16(qv);
    kp[(size_t)t * (NHEAD * KPADH) + x] = __float2bfloat16(kv);
  }
  for (int x = threadIdx.x; x < NHEAD * 80; x += blockDim.x) {
    int h = x / 80, f = x % 80;
    float v = (f < KHEAD) ? b2f(row[2 * DMODEL + h * KHEAD + f]) : 0.f;
    vT[(size_t)(h * 80 + f) * NTOK + t] = __float2bfloat16(v);
  }
}

// ---------------- WMMA flash-ish attention: one wave per (16-query tile, head)
__global__ void attn_kernel(const bfst* __restrict__ qp, const bfst* __restrict__ kp,
                            const bfst* __restrict__ vT, bfst* __restrict__ attn_out) {
  extern __shared__ char smem[];
  bfst* P = reinterpret_cast<bfst*>(smem);      // [16][n+PPAD] scores/probs (bf16)
  __shared__ float rs[16];
  int tile = blockIdx.x, h = blockIdx.y, lane = threadIdx.x;
  int segStart, n, qbase;
  if (tile < 144)      { segStart = 0;    n = 2304; qbase = tile * 16; }
  else if (tile < 272) { segStart = 2304; n = 2048; qbase = 2304 + (tile - 144) * 16; }
  else if (tile < 332) { segStart = 4352; n = 960;  qbase = 4352 + (tile - 272) * 16; }
  else                 { segStart = 5312; n = 960;  qbase = 5312 + (tile - 332) * 16; }
  int ldp = n + PPAD;
  int m16 = lane & 15;
  int c8 = (lane >> 4) * 8;
  int mo = (lane >> 4) * 8;

  // Q fragments: A-matrix 16x32 layout x3 (head dim padded to 96)
  v16bf qa[3];
  const bfst* qrow = qp + (size_t)(qbase + m16) * (NHEAD * KPADH) + h * KPADH;
#pragma unroll
  for (int kc = 0; kc < 3; ++kc)
    qa[kc] = mk16(ld8(qrow + kc * 32 + c8), ld8(qrow + kc * 32 + 16 + c8));

  // phase 1: S = Q @ K^T, stored bf16 in LDS
  for (int kt = 0; kt < n / 16; ++kt) {
    const bfst* krow = kp + (size_t)(segStart + kt * 16 + m16) * (NHEAD * KPADH)
                          + h * KPADH;
    v8f acc;
#pragma unroll
    for (int i = 0; i < 8; ++i) acc[i] = 0.f;
#pragma unroll
    for (int kc = 0; kc < 3; ++kc)
      acc = wmma_bf(qa[kc], mk16(ld8(krow + kc * 32 + c8),
                                 ld8(krow + kc * 32 + 16 + c8)), acc);
#pragma unroll
    for (int r = 0; r < 8; ++r)
      P[(mo + r) * ldp + kt * 16 + m16] = __float2bfloat16(acc[r]);
  }
  __syncthreads();

  // phase 2: in-place softmax, whole wave cooperates on each row
  for (int r = 0; r < 16; ++r) {
    bfst* prow = P + r * ldp;
    float mx = -1e30f;
    for (int j = lane; j < n; j += 32) mx = fmaxf(mx, b2f(prow[j]));
    mx = wmax32(mx);
    float sum = 0.f;
    for (int j = lane; j < n; j += 32) {
      float e = __expf(b2f(prow[j]) - mx);
      sum += e;
      prow[j] = __float2bfloat16(e);
    }
    sum = wred32(sum);
    if (lane == 0) rs[r] = 1.f / sum;
  }
  __syncthreads();

  // phase 3: O = P @ V (V pre-transposed, feature rows contiguous over tokens)
  v8f o[5];
#pragma unroll
  for (int ft = 0; ft < 5; ++ft)
#pragma unroll
    for (int i = 0; i < 8; ++i) o[ft][i] = 0.f;
  for (int kq = 0; kq < n / 32; ++kq) {
    const bfst* pr = P + m16 * ldp + kq * 32;
    v16bf pa = mk16(*reinterpret_cast<const v8bf*>(pr + c8),
                    *reinterpret_cast<const v8bf*>(pr + 16 + c8));
#pragma unroll
    for (int ft = 0; ft < 5; ++ft) {
      const bfst* vrow = vT + (size_t)(h * 80 + ft * 16 + m16) * NTOK
                            + segStart + kq * 32;
      o[ft] = wmma_bf(pa, mk16(ld8(vrow + c8), ld8(vrow + 16 + c8)), o[ft]);
    }
  }
#pragma unroll
  for (int ft = 0; ft < 5; ++ft) {
    int fn = ft * 16 + m16;
    if (fn >= KHEAD) continue;
#pragma unroll
    for (int r = 0; r < 8; ++r) {
      int M = mo + r;
      attn_out[(size_t)(qbase + M) * DMODEL + h * KHEAD + fn] =
          __float2bfloat16(o[ft][r] * rs[M]);
    }
  }
}

// ---------------- generic WMMA GEMM: C = A(bf16) @ Wt^T + bias ----------------
// block = 128 threads (4 waves); block tile 128(M) x 64(N); wave owns 32 rows.
// B tile staged to LDS via gfx1250 async global->LDS DMA (ASYNCcnt).
__global__ void gemm_kernel(const bfst* __restrict__ A, int lda,
                            const bfst* __restrict__ Wt, int Kpad,
                            const float* __restrict__ bias,
                            const float* __restrict__ resid,
                            float* __restrict__ f32_out, int ldo,
                            bfst* __restrict__ bf_out, int ldbf,
                            int Mrows, int N, int doGelu) {
  __shared__ bfst Bt[64 * 32];
  int w = threadIdx.x >> 5, lane = threadIdx.x & 31;
  int m16 = lane & 15, c8 = (lane >> 4) * 8;
  int rb = blockIdx.x * 128 + w * 32;
  int n0 = blockIdx.y * 64;
  int row0 = rb + m16, row1 = rb + 16 + m16;
  bool rok0 = row0 < Mrows, rok1 = row1 < Mrows;
  const bfst* arow0 = A + (size_t)row0 * lda;
  const bfst* arow1 = A + (size_t)row1 * lda;

  v8f acc[2][4];
#pragma unroll
  for (int s = 0; s < 2; ++s)
#pragma unroll
    for (int j = 0; j < 4; ++j)
#pragma unroll
      for (int i = 0; i < 8; ++i) acc[s][j][i] = 0.f;

  int nn = threadIdx.x >> 1, kk = (threadIdx.x & 1) * 16;
  for (int k0 = 0; k0 < Kpad; k0 += 32) {
    __syncthreads();
    // async DMA of B tile (64 cols x 32 k) straight into LDS
    async_g2l_32B(Wt + (size_t)(n0 + nn) * Kpad + k0 + kk, Bt + nn * 32 + kk);
    wait_async0();
    __syncthreads();

    v16bf a0, a1;
#pragma unroll
    for (int i = 0; i < 16; ++i) { a0[i] = zbf(); a1[i] = zbf(); }
    if (rok0) {
      a0 = mk16(ld8(arow0 + k0 + c8), ld8(arow0 + k0 + 16 + c8));
      if (k0 + 32 < Kpad) __builtin_prefetch(arow0 + k0 + 32, 0, 1);
    }
    if (rok1) {
      a1 = mk16(ld8(arow1 + k0 + c8), ld8(arow1 + k0 + 16 + c8));
      if (k0 + 32 < Kpad) __builtin_prefetch(arow1 + k0 + 32, 0, 1);
    }
#pragma unroll
    for (int j = 0; j < 4; ++j) {
      const bfst* bp = Bt + (j * 16 + m16) * 32;
      v16bf b = mk16(*reinterpret_cast<const v8bf*>(bp + c8),
                     *reinterpret_cast<const v8bf*>(bp + 16 + c8));
      acc[0][j] = wmma_bf(a0, b, acc[0][j]);
      acc[1][j] = wmma_bf(a1, b, acc[1][j]);
    }
  }

  // epilogue: bias + gelu + residual, f32 and/or bf16 (zero in N-pad region)
#pragma unroll
  for (int s = 0; s < 2; ++s) {
#pragma unroll
    for (int j = 0; j < 4; ++j) {
      int gn = n0 + j * 16 + m16;
      bool nok = gn < N;
#pragma unroll
      for (int r = 0; r < 8; ++r) {
        int gm = rb + s * 16 + r + ((lane >> 4) << 3);
        if (gm >= Mrows) continue;
        float val = acc[s][j][r];
        if (nok && bias) val += bias[gn];
        if (doGelu) val = gelu_f(val);
        if (resid && nok) val += resid[(size_t)gm * ldo + gn];
        if (f32_out && nok) f32_out[(size_t)gm * ldo + gn] = val;
        if (bf_out) bf_out[(size_t)gm * ldbf + gn] = __float2bfloat16(nok ? val : 0.f);
      }
    }
  }
}

// ---------------- host side ----------------
extern "C" void kernel_launch(void* const* d_in, const int* in_sizes, int n_in,
                              void* d_out, int out_size, void* d_ws, size_t ws_size,
                              hipStream_t stream) {
  (void)in_sizes; (void)n_in; (void)out_size; (void)ws_size;
  const float* pixel   = (const float*)d_in[0];
  const float* patch_w = (const float*)d_in[2];
  const float* patch_b = (const float*)d_in[3];
  const float* pos_tab = (const float*)d_in[4];
  const float* ln1_s   = (const float*)d_in[5];
  const float* ln1_b   = (const float*)d_in[6];
  const float* qkv_w   = (const float*)d_in[7];
  const float* qkv_b   = (const float*)d_in[8];
  const float* proj_w  = (const float*)d_in[9];
  const float* proj_b  = (const float*)d_in[10];
  const float* ln2_s   = (const float*)d_in[11];
  const float* ln2_b   = (const float*)d_in[12];
  const float* fc1_w   = (const float*)d_in[13];
  const float* fc1_b   = (const float*)d_in[14];
  const float* fc2_w   = (const float*)d_in[15];
  const float* fc2_b   = (const float*)d_in[16];
  const float* mln_s   = (const float*)d_in[17];
  const float* mln_b   = (const float*)d_in[18];
  const float* mfc1_w  = (const float*)d_in[19];
  const float* mfc1_b  = (const float*)d_in[20];
  const float* mfc2_w  = (const float*)d_in[21];
  const float* mfc2_b  = (const float*)d_in[22];
  float* out = (float*)d_out;

  // workspace carve
  char* base = (char*)d_ws;
  size_t off = 0;
  auto alloc = [&](size_t bytes) -> void* {
    void* p = base + off;
    off += (bytes + 255) & ~(size_t)255;
    return p;
  };
  bfst* bfA  = (bfst*)alloc((size_t)NTOK * MDIM * 2);
  bfst* bfB  = (bfst*)alloc((size_t)NTOK * MDIM * 2);
  float* X   = (float*)alloc((size_t)NTOK * DMODEL * 4);
  bfst* qp   = (bfst*)alloc((size_t)NTOK * NHEAD * KPADH * 2);
  bfst* kp   = (bfst*)alloc((size_t)NTOK * NHEAD * KPADH * 2);
  bfst* vT   = (bfst*)alloc((size_t)NHEAD * 80 * NTOK * 2);
  float* cosT = (float*)alloc((size_t)NTOK * KHEAD * 4);
  float* sinT = (float*)alloc((size_t)NTOK * KHEAD * 4);
  bfst* Wt   = (bfst*)alloc((size_t)MDIM * MDIM * 2);

  auto wconv = [&](const float* W, int Kd, int N, int Kpad) {
    size_t tot = (size_t)N * Kpad;
    wt_kernel<<<(unsigned)((tot + 255) / 256), 256, 0, stream>>>(W, Wt, Kd, N, Kpad);
  };
  auto gemm = [&](const bfst* A, int lda, int Kpad, const float* bias,
                  const float* resid, float* fo, int ldo, bfst* bo, int ldbf,
                  int Mr, int N, int Npad, int gelu) {
    dim3 g((Mr + 127) / 128, Npad / 64);
    gemm_kernel<<<g, 128, 0, stream>>>(A, lda, Wt, Kpad, bias, resid, fo, ldo, bo,
                                       ldbf, Mr, N, gelu);
  };

  // patch embed: X = pixel @ patch_w + patch_b
  f2bf_kernel<<<(unsigned)(((size_t)NTOK * CIN + 255) / 256), 256, 0, stream>>>(
      pixel, bfA, (size_t)NTOK * CIN);
  wconv(patch_w, CIN, DMODEL, CIN);
  gemm(bfA, CIN, CIN, patch_b, nullptr, X, DMODEL, nullptr, 0, NTOK, DMODEL, DMODEL, 0);
  posadd_kernel<<<NTOK, 128, 0, stream>>>(pos_tab, X);
  rope_tab_kernel<<<NTOK, 64, 0, stream>>>(cosT, sinT);

  for (int l = 0; l < 2; ++l) {
    // LN1 -> qkv GEMM
    ln_kernel<<<NTOK / 4, 128, 0, stream>>>(X, ln1_s + l * DMODEL, ln1_b + l * DMODEL,
                                            bfA, DMODEL);
    wconv(qkv_w + (size_t)l * DMODEL * QKVD, DMODEL, QKVD, DMODEL);
    gemm(bfA, DMODEL, DMODEL, qkv_b + (size_t)l * QKVD, nullptr, nullptr, 0, bfB,
         QKVD, NTOK, QKVD, QKVD, 0);
    // rope + v transpose
    rope_apply_kernel<<<NTOK, 128, 0, stream>>>(bfB, cosT, sinT, qp, kp, vT);
    // attention -> bfA
    attn_kernel<<<dim3(392, NHEAD), 32, 16 * (2304 + PPAD) * 2, stream>>>(qp, kp, vT,
                                                                          bfA);
    // proj + residual
    wconv(proj_w + (size_t)l * DMODEL * DMODEL, DMODEL, DMODEL, DMODEL);
    gemm(bfA, DMODEL, DMODEL, proj_b + (size_t)l * DMODEL, X, X, DMODEL, nullptr, 0,
         NTOK, DMODEL, DMODEL, 0);
    // LN2 -> MLP
    ln_kernel<<<NTOK / 4, 128, 0, stream>>>(X, ln2_s + l * DMODEL, ln2_b + l * DMODEL,
                                            bfA, DMODEL);
    wconv(fc1_w + (size_t)l * DMODEL * IDIM, DMODEL, IDIM, DMODEL);
    gemm(bfA, DMODEL, DMODEL, fc1_b + (size_t)l * IDIM, nullptr, nullptr, 0, bfB,
         IPAD, NTOK, IDIM, IPAD, 1);
    wconv(fc2_w + (size_t)l * IDIM * DMODEL, IDIM, DMODEL, IPAD);
    gemm(bfB, IPAD, IPAD, fc2_b + (size_t)l * DMODEL, X, X, DMODEL, nullptr, 0,
         NTOK, DMODEL, DMODEL, 0);
  }

  // merger: LN -> (1568 x 4608) view -> fc1 (gelu) -> fc2
  ln_kernel<<<NTOK / 4, 128, 0, stream>>>(X, mln_s, mln_b, bfA, DMODEL);
  wconv(mfc1_w, MDIM, MDIM, MDIM);
  gemm(bfA, MDIM, MDIM, mfc1_b, nullptr, nullptr, 0, bfB, MDIM, NTOK / 4, MDIM,
       MDIM, 1);
  wconv(mfc2_w, MDIM, OUTD, MDIM);
  gemm(bfB, MDIM, MDIM, mfc2_b, nullptr, out, OUTD, nullptr, 0, NTOK / 4, OUTD,
       OUTD, 0);
}
